// MLAttention_66786741453102
// MI455X (gfx1250) — matrix operand order repaired and checked by reading, not compiled, not run on hardware
//
#include <hip/hip_runtime.h>
#include <hip/hip_bf16.h>

typedef __attribute__((ext_vector_type(16))) _Float16 v16h;
typedef __attribute__((ext_vector_type(8)))  _Float16 v8h;
typedef __attribute__((ext_vector_type(4)))  _Float16 v4h;
typedef __attribute__((ext_vector_type(8)))  float    v8f;
typedef __attribute__((ext_vector_type(4)))  unsigned int u32x4;
typedef __attribute__((ext_vector_type(8)))  int      i32x8;
typedef __attribute__((ext_vector_type(4)))  int      i32x4;

#define NH_     16
#define NOPE_D_ 48
#define ROPE_D_ 16
#define V_D_    64
#define R_      128
#define HD_     64          // NOPE_D + ROPE_D
#define D_      1024
#define T_      2048
#define B_      2
#define EPS_    1e-6f

#if defined(__has_builtin)
#if __has_builtin(__builtin_amdgcn_tensor_load_to_lds)
#define HAVE_TDM 1
#endif
#endif

#define WMMA_F16(a, b, c) \
  __builtin_amdgcn_wmma_f32_16x16x32_f16(false, (a), false, (b), (short)0, (c), false, false)

// ---------------------------------------------------------------------------
// Fragment loaders (wave32 layouts per CDNA5 ISA 7.12.2)
// A 16x32 f16: lane l -> row (l&15); K chunks {0..7,16..23} (l<16) else {8..15,24..31}
// B 32x16 f16: lane l -> col (l&15); K chunks {0..7,16..23} (l<16) else {8..15,24..31}
// Strides are multiples of 8 halves so every chunk is a 16B-aligned ds_load_b128.
// ---------------------------------------------------------------------------
__device__ __forceinline__ v16h load_a_frag(const _Float16* lds, int stride, int row0, int kbase) {
  int lane = threadIdx.x & 31;
  const _Float16* p = lds + (size_t)(row0 + (lane & 15)) * stride + kbase + ((lane < 16) ? 0 : 8);
  v8h c0 = *(const v8h*)(p);
  v8h c1 = *(const v8h*)(p + 16);
  v16h a;
#pragma unroll
  for (int i = 0; i < 8; ++i) { a[i] = c0[i]; a[i + 8] = c1[i]; }
  return a;
}

__device__ __forceinline__ v16h load_b_frag(const _Float16* lds, int stride, int row0, int kbase) {
  int lane = threadIdx.x & 31;
  const _Float16* p = lds + (size_t)(row0 + (lane & 15)) * stride + kbase + ((lane < 16) ? 0 : 8);
  v8h c0 = *(const v8h*)(p);
  v8h c1 = *(const v8h*)(p + 16);
  v16h b;
#pragma unroll
  for (int i = 0; i < 8; ++i) { b[i] = c0[i]; b[i + 8] = c1[i]; }
  return b;
}

__device__ __forceinline__ v4h cvt4(float a, float b, float c, float d) {
  v4h h;
  h[0] = (_Float16)a; h[1] = (_Float16)b; h[2] = (_Float16)c; h[3] = (_Float16)d;
  return h;
}

// ---------------------------------------------------------------------------
// Generic GEMM: C(M,N) f32 = A(M,K) f32 * W(N,K)^T f32, f16 WMMA / f32 acc.
// Block tile 128x64, Kc=64, double-buffered LDS, software-pipelined:
//   load_regs(t+1)  ->  compute(t)  ->  store_lds(t+1)  ->  barrier
// so global loads for the next tile fly while WMMAs for this tile execute.
// 8 waves (4M x 2N), each wave owns a 32x32 sub-tile (2x2 frags, 8 WMMA/step).
// ---------------------------------------------------------------------------
#define GBM 128
#define GBN 64
#define GBK 64
#define GAS 72   // LDS stride in halves (mult of 8 -> 16B aligned chunks)

__global__ __launch_bounds__(256) void gemm_f32_wmma(
    const float* __restrict__ A, const float* __restrict__ W,
    float* __restrict__ C, int M, int N, int K) {
  __shared__ __align__(16) _Float16 As[2][GBM * GAS];
  __shared__ __align__(16) _Float16 Ws[2][GBN * GAS];

  const int tid = threadIdx.x;
  const int wave = tid >> 5, lane = tid & 31;
  const int wm = wave >> 1, wn = wave & 1;      // 4x2 wave grid
  const int m0 = blockIdx.x * GBM;
  const int n0 = blockIdx.y * GBN;

  v8f acc[2][2] = {};
  float4 ra[8];   // staged A quads (batched loads -> overlapped latency)
  float4 rw[4];   // staged W quads

  auto load_regs = [&](int k0) {
#pragma unroll
    for (int j = 0; j < 8; ++j) {
      int i = tid + j * 256;
      int r = i >> 4, c4 = (i & 15) * 4;
      ra[j] = *(const float4*)(A + (size_t)(m0 + r) * K + (k0 + c4));
    }
#pragma unroll
    for (int j = 0; j < 4; ++j) {
      int i = tid + j * 256;
      int r = i >> 4, c4 = (i & 15) * 4;
      rw[j] = make_float4(0.f, 0.f, 0.f, 0.f);
      if (n0 + r < N) rw[j] = *(const float4*)(W + (size_t)(n0 + r) * K + (k0 + c4));
    }
    // One prefetch per thread covers the K-step after next (128B granularity).
    if (k0 + GBK < K)
      __builtin_prefetch(A + (size_t)(m0 + (tid >> 1)) * K + (k0 + GBK) + (tid & 1) * 32, 0, 0);
  };

  auto store_lds = [&](int p) {
#pragma unroll
    for (int j = 0; j < 8; ++j) {
      int i = tid + j * 256;
      int r = i >> 4, c4 = (i & 15) * 4;
      *(v4h*)(&As[p][r * GAS + c4]) = cvt4(ra[j].x, ra[j].y, ra[j].z, ra[j].w);
    }
#pragma unroll
    for (int j = 0; j < 4; ++j) {
      int i = tid + j * 256;
      int r = i >> 4, c4 = (i & 15) * 4;
      *(v4h*)(&Ws[p][r * GAS + c4]) = cvt4(rw[j].x, rw[j].y, rw[j].z, rw[j].w);
    }
  };

  auto compute = [&](int p) {
#pragma unroll
    for (int kk = 0; kk < GBK; kk += 32) {
      v16h a0 = load_a_frag(As[p], GAS, wm * 32, kk);
      v16h a1 = load_a_frag(As[p], GAS, wm * 32 + 16, kk);
      v16h b0 = load_b_frag(Ws[p], GAS, wn * 32, kk);
      v16h b1 = load_b_frag(Ws[p], GAS, wn * 32 + 16, kk);
      acc[0][0] = WMMA_F16(a0, b0, acc[0][0]);
      acc[0][1] = WMMA_F16(a0, b1, acc[0][1]);
      acc[1][0] = WMMA_F16(a1, b0, acc[1][0]);
      acc[1][1] = WMMA_F16(a1, b1, acc[1][1]);
    }
  };

  const int nk = K / GBK;
  load_regs(0);
  store_lds(0);
  __syncthreads();
  for (int t = 0; t < nk; ++t) {
    const int p = t & 1;
    if (t + 1 < nk) load_regs((t + 1) * GBK);   // global loads in flight...
    compute(p);                                 // ...while WMMAs run
    if (t + 1 < nk) store_lds(p ^ 1);
    __syncthreads();
  }

  // Store: C layout lane l -> col (l&15); VGPR r -> row r (+8 for lanes 16..31)
  const int rbase = (lane < 16) ? 0 : 8;
#pragma unroll
  for (int mi = 0; mi < 2; ++mi) {
#pragma unroll
    for (int ni = 0; ni < 2; ++ni) {
      int col = n0 + wn * 32 + ni * 16 + (lane & 15);
      if (col < N) {
        int row = m0 + wm * 32 + mi * 16 + rbase;
#pragma unroll
        for (int r = 0; r < 8; ++r)
          C[(size_t)(row + r) * N + col] = acc[mi][ni][r];
      }
    }
  }
}

// ---------------------------------------------------------------------------
// Q head RMSNorm + RoPE, f32 -> f16  (one thread per (b,t,h))
// ---------------------------------------------------------------------------
__global__ __launch_bounds__(256) void q_norm_rope(
    const float* __restrict__ q_raw, const float* __restrict__ qw,
    const float* __restrict__ cosb, const float* __restrict__ sinb,
    _Float16* __restrict__ qf) {
  int idx = blockIdx.x * blockDim.x + threadIdx.x;
  if (idx >= B_ * T_ * NH_) return;
  int h = idx % NH_;
  int bt = idx / NH_;
  int t = bt % T_;

  const float* src = q_raw + (size_t)bt * (NH_ * HD_) + h * HD_;
  float v[HD_];
  float ss = 0.f;
#pragma unroll
  for (int i = 0; i < HD_; i += 4) {
    float4 f = *(const float4*)(src + i);
    v[i] = f.x; v[i + 1] = f.y; v[i + 2] = f.z; v[i + 3] = f.w;
    ss += f.x * f.x + f.y * f.y + f.z * f.z + f.w * f.w;
  }
  float inv = rsqrtf(ss * (1.0f / HD_) + EPS_);
#pragma unroll
  for (int i = 0; i < HD_; ++i) v[i] *= inv * qw[i];

  _Float16* dst = qf + (size_t)idx * HD_;
#pragma unroll
  for (int i = 0; i < NOPE_D_; i += 4)
    *(v4h*)(dst + i) = cvt4(v[i], v[i + 1], v[i + 2], v[i + 3]);
#pragma unroll
  for (int i = 0; i < 8; ++i) {
    float c = cosb[t * 8 + i], sn = sinb[t * 8 + i];
    float x1 = v[NOPE_D_ + i], x2 = v[NOPE_D_ + 8 + i];
    dst[NOPE_D_ + i]     = (_Float16)(x1 * c - x2 * sn);
    dst[NOPE_D_ + 8 + i] = (_Float16)(x2 * c + x1 * sn);
  }
}

// ---------------------------------------------------------------------------
// kv_a split: RMSNorm(latent 128) f32 out, RoPE(k_rope 16) f32 out
// ---------------------------------------------------------------------------
__global__ __launch_bounds__(256) void kva_process(
    const float* __restrict__ kv_a, const float* __restrict__ kw,
    const float* __restrict__ cosb, const float* __restrict__ sinb,
    float* __restrict__ latent, float* __restrict__ krope) {
  int bt = blockIdx.x * blockDim.x + threadIdx.x;
  if (bt >= B_ * T_) return;
  int t = bt % T_;
  const float* src = kv_a + (size_t)bt * (R_ + ROPE_D_);
  float ss = 0.f;
#pragma unroll 8
  for (int i = 0; i < R_; ++i) ss += src[i] * src[i];
  float inv = rsqrtf(ss * (1.0f / R_) + EPS_);
  float* lat = latent + (size_t)bt * R_;
#pragma unroll 8
  for (int i = 0; i < R_; ++i) lat[i] = src[i] * inv * kw[i];
  float* kr = krope + (size_t)bt * ROPE_D_;
#pragma unroll
  for (int i = 0; i < 8; ++i) {
    float c = cosb[t * 8 + i], sn = sinb[t * 8 + i];
    float x1 = src[R_ + i], x2 = src[R_ + 8 + i];
    kr[i]     = x1 * c - x2 * sn;
    kr[8 + i] = x2 * c + x1 * sn;
  }
}

// ---------------------------------------------------------------------------
// Pack kv_b (f32) + k_rope -> kf, vf (f16, (b,t,h,64))
// ---------------------------------------------------------------------------
__global__ __launch_bounds__(256) void pack_kv(
    const float* __restrict__ kv_b, const float* __restrict__ krope,
    _Float16* __restrict__ kf, _Float16* __restrict__ vf) {
  int idx = blockIdx.x * blockDim.x + threadIdx.x;
  if (idx >= B_ * T_ * NH_) return;
  int h = idx % NH_;
  int bt = idx / NH_;
  const float* src = kv_b + (size_t)bt * (NH_ * (NOPE_D_ + V_D_)) + h * (NOPE_D_ + V_D_);
  _Float16* kd = kf + (size_t)idx * HD_;
  _Float16* vd = vf + (size_t)idx * V_D_;
#pragma unroll
  for (int i = 0; i < NOPE_D_; i += 4)
    *(v4h*)(kd + i) = cvt4(src[i], src[i + 1], src[i + 2], src[i + 3]);
  const float* kr = krope + (size_t)bt * ROPE_D_;
#pragma unroll
  for (int i = 0; i < ROPE_D_; i += 4)
    *(v4h*)(kd + NOPE_D_ + i) = cvt4(kr[i], kr[i + 1], kr[i + 2], kr[i + 3]);
#pragma unroll
  for (int i = 0; i < V_D_; i += 4)
    *(v4h*)(vd + i) = cvt4(src[NOPE_D_ + i], src[NOPE_D_ + i + 1],
                           src[NOPE_D_ + i + 2], src[NOPE_D_ + i + 3]);
}

// ---------------------------------------------------------------------------
// Flash attention (causal): block = (q-tile 64, head, batch), 4 waves, each
// wave owns 16 query rows. K-tile staged by the Tensor Data Mover (TDM),
// D# built with pad_enable so rows land at the padded LDS stride. V staged
// transposed (batched loads then scatter). QK^T and PV via WMMA; online
// softmax with half-wave shuffle reductions.
// ---------------------------------------------------------------------------
#define QT_ 64
#define KT_ 64
#define AST_ 72   // LDS stride halves (mult of 8)

__global__ __launch_bounds__(128) void mla_flash_attn(
    const _Float16* __restrict__ qf, const _Float16* __restrict__ kf,
    const _Float16* __restrict__ vf, float* __restrict__ attn_out) {
  __shared__ __align__(16) _Float16 Qs[QT_ * AST_];
  __shared__ __align__(16) _Float16 Ks[KT_ * AST_];
  __shared__ __align__(16) _Float16 Vts[V_D_ * AST_];   // transposed: [d][key]
  __shared__ __align__(16) _Float16 Ps[4 * 16 * AST_];  // per-wave P strips

  const int tid = threadIdx.x;
  const int wave = tid >> 5, lane = tid & 31;
  const int qt = blockIdx.x, h = blockIdx.y, b = blockIdx.z;
  const int q0 = qt * QT_;
  const int rbase = (lane < 16) ? 0 : 8;
  const float scale = 0.125f;   // 64^-0.5

  // Stage Q tile (64 rows x 64 d) with v8h (b128) transfers: 4 iters/thread
#pragma unroll
  for (int i = tid; i < (QT_ * HD_) / 8; i += 128) {
    int r = i >> 3, d8 = (i & 7) * 8;
    *(v8h*)(Qs + r * AST_ + d8) =
        *(const v8h*)(qf + (((size_t)(b * T_ + q0 + r)) * NH_ + h) * HD_ + d8);
  }
  __syncthreads();

  v16h aq0 = load_a_frag(Qs, AST_, wave * 16, 0);
  v16h aq1 = load_a_frag(Qs, AST_, wave * 16, 32);

  float mrow[8], lrow[8];
  v8f o[4] = {};
#pragma unroll
  for (int r = 0; r < 8; ++r) { mrow[r] = -1e30f; lrow[r] = 0.0f; }

  for (int kb = 0; kb <= q0; kb += KT_) {
    __syncthreads();

#ifdef HAVE_TDM
    // --- Tensor Data Mover: 2D tile 64x64 f16, row stride NH*64 halves,
    //     pad 4 DWORDs after every 32 DWORDs -> LDS stride 72 halves.
    if (wave == 0) {
      unsigned long long ga =
          (unsigned long long)(uintptr_t)(kf + (((size_t)(b * T_ + kb)) * NH_ + h) * HD_);
      unsigned int la = (unsigned int)(uintptr_t)(void*)Ks;
      u32x4 g0;
      g0[0] = 1u;                                    // count=1, user descriptor
      g0[1] = la;                                    // lds_addr
      g0[2] = (unsigned int)ga;                      // global_addr[31:0]
      g0[3] = (unsigned int)((ga >> 32) & 0x01FFFFFFu) | 0x80000000u; // addr[56:32] | type=2
      i32x8 g1;
      g1[0] = (1 << 16) | (1 << 20) | (4 << 22) | (3 << 25); // 2B elems, pad_en, int=32dw, amt=4dw
      g1[1] = (int)(64u << 16);                      // tensor_dim0 = 64 (low 16)
      g1[2] = (int)(((unsigned)(T_ - kb) & 0xFFFFu) << 16);  // tensor_dim1 low16
      g1[3] = (int)(64u << 16);                      // tile_dim0 = 64
      g1[4] = 64;                                    // tile_dim1 = 64
      g1[5] = NH_ * HD_;                             // tensor_dim0_stride = 1024
      g1[6] = 0; g1[7] = 0;
      i32x4 gz4 = {0, 0, 0, 0};
      i32x8 gz8 = {0, 0, 0, 0, 0, 0, 0, 0};
      // 6-arg form (clang-23 / therock-10.0 toolchain)
      __builtin_amdgcn_tensor_load_to_lds(g0, g1, gz4, gz4, gz8, 0);
    }
#else
    // Fallback: vectorized manual staging of K tile
#pragma unroll
    for (int i = tid; i < (KT_ * HD_) / 8; i += 128) {
      int r = i >> 3, d8 = (i & 7) * 8;
      *(v8h*)(Ks + r * AST_ + d8) =
          *(const v8h*)(kf + (((size_t)(b * T_ + kb + r)) * NH_ + h) * HD_ + d8);
    }
#endif

    // Stage V transposed ([d][key]): batch the 4 v8h loads, then scatter b16
    {
      v8h rv[4];
#pragma unroll
      for (int j = 0; j < 4; ++j) {
        int i = tid + j * 128;
        int key = i >> 3, d8 = (i & 7) * 8;
        rv[j] = *(const v8h*)(vf + (((size_t)(b * T_ + kb + key)) * NH_ + h) * V_D_ + d8);
      }
#pragma unroll
      for (int j = 0; j < 4; ++j) {
        int i = tid + j * 128;
        int key = i >> 3, d8 = (i & 7) * 8;
#pragma unroll
        for (int jj = 0; jj < 8; ++jj) Vts[(d8 + jj) * AST_ + key] = rv[j][jj];
      }
    }

#ifdef HAVE_TDM
    if (wave == 0) __builtin_amdgcn_s_wait_tensorcnt(0);
#endif
    __syncthreads();

    // S = Q K^T  (16 q-rows x 64 keys per wave)
    v8f s[4];
#pragma unroll
    for (int nf = 0; nf < 4; ++nf) {
      v16h bk0 = load_b_frag(Ks, AST_, nf * 16, 0);
      v16h bk1 = load_b_frag(Ks, AST_, nf * 16, 32);
      v8f z = {};
      z = WMMA_F16(aq0, bk0, z);
      z = WMMA_F16(aq1, bk1, z);
      s[nf] = z;
    }

    // Scale + causal mask (only diagonal tile needs it)
    const bool diag = (kb == q0);
#pragma unroll
    for (int nf = 0; nf < 4; ++nf) {
      int kj = kb + nf * 16 + (lane & 15);
#pragma unroll
      for (int r = 0; r < 8; ++r) {
        float v = s[nf][r] * scale;
        if (diag && kj > q0 + wave * 16 + rbase + r) v = -1e30f;
        s[nf][r] = v;
      }
    }

    // Row max across 64 keys: frag-local max, then xor-shuffles in 16-lane group
    float rm[8];
#pragma unroll
    for (int r = 0; r < 8; ++r) {
      float x = fmaxf(fmaxf(s[0][r], s[1][r]), fmaxf(s[2][r], s[3][r]));
      x = fmaxf(x, __shfl_xor(x, 1, 32));
      x = fmaxf(x, __shfl_xor(x, 2, 32));
      x = fmaxf(x, __shfl_xor(x, 4, 32));
      x = fmaxf(x, __shfl_xor(x, 8, 32));
      rm[r] = x;
    }

    // Online softmax update
#pragma unroll
    for (int r = 0; r < 8; ++r) {
      float mnew = fmaxf(mrow[r], rm[r]);
      float alpha = __expf(mrow[r] - mnew);
      mrow[r] = mnew;
      float psum = 0.f;
#pragma unroll
      for (int nf = 0; nf < 4; ++nf) {
        float p = __expf(s[nf][r] - mnew);
        s[nf][r] = p;
        psum += p;
      }
      psum += __shfl_xor(psum, 1, 32);
      psum += __shfl_xor(psum, 2, 32);
      psum += __shfl_xor(psum, 4, 32);
      psum += __shfl_xor(psum, 8, 32);
      lrow[r] = lrow[r] * alpha + psum;
#pragma unroll
      for (int nf = 0; nf < 4; ++nf) o[nf][r] *= alpha;
    }

    // Spill P (C-layout) to LDS so it can be reloaded as A-fragments
#pragma unroll
    for (int nf = 0; nf < 4; ++nf) {
      int col = nf * 16 + (lane & 15);
#pragma unroll
      for (int r = 0; r < 8; ++r)
        Ps[(size_t)(wave * 16 + rbase + r) * AST_ + col] = (_Float16)s[nf][r];
    }
    __syncthreads();

    // O += P V
    v16h ap0 = load_a_frag(Ps, AST_, wave * 16, 0);
    v16h ap1 = load_a_frag(Ps, AST_, wave * 16, 32);
#pragma unroll
    for (int nf = 0; nf < 4; ++nf) {
      v16h bv0 = load_b_frag(Vts, AST_, nf * 16, 0);
      v16h bv1 = load_b_frag(Vts, AST_, nf * 16, 32);
      o[nf] = WMMA_F16(ap0, bv0, o[nf]);
      o[nf] = WMMA_F16(ap1, bv1, o[nf]);
    }
  }

  // Epilogue: normalize and write (b, q, h*64 + d) f32
#pragma unroll
  for (int nf = 0; nf < 4; ++nf) {
    int col = h * V_D_ + nf * 16 + (lane & 15);
#pragma unroll
    for (int r = 0; r < 8; ++r) {
      int q = q0 + wave * 16 + rbase + r;
      attn_out[((size_t)(b * T_ + q)) * (NH_ * V_D_) + col] = o[nf][r] / lrow[r];
    }
  }
}

// ---------------------------------------------------------------------------
// Host orchestration
// ---------------------------------------------------------------------------
extern "C" void kernel_launch(void* const* d_in, const int* in_sizes, int n_in,
                              void* d_out, int out_size, void* d_ws, size_t ws_size,
                              hipStream_t stream) {
  (void)in_sizes; (void)n_in; (void)out_size; (void)ws_size;
  const float* x        = (const float*)d_in[0];   // (B,T,D)
  const float* cosb     = (const float*)d_in[1];   // (T,8)
  const float* sinb     = (const float*)d_in[2];   // (T,8)
  const float* Wq       = (const float*)d_in[3];   // (NH*64, D)
  const float* q_norm_w = (const float*)d_in[4];   // (64,)
  const float* Wkva     = (const float*)d_in[5];   // (144, D)
  const float* kva_w    = (const float*)d_in[6];   // (128,)
  const float* Wkvb     = (const float*)d_in[7];   // (1792, 128)
  const float* Wo       = (const float*)d_in[8];   // (D, 1024)
  float* out = (float*)d_out;

  const int M = B_ * T_;           // 4096
  char* ws = (char*)d_ws;
  size_t off = 0;
  auto alloc = [&](size_t bytes) { size_t o = off; off += (bytes + 255) & ~(size_t)255; return o; };

  float*    q_raw  = (float*)(ws + alloc((size_t)M * (NH_ * HD_) * 4));        // 16 MB
  float*    kv_a   = (float*)(ws + alloc((size_t)M * (R_ + ROPE_D_) * 4));     // 2.25 MB
  float*    latent = (float*)(ws + alloc((size_t)M * R_ * 4));                 // 2 MB
  float*    krope  = (float*)(ws + alloc((size_t)M * ROPE_D_ * 4));            // 0.25 MB
  float*    kv_b   = (float*)(ws + alloc((size_t)M * NH_ * (NOPE_D_ + V_D_) * 4)); // 28 MB
  _Float16* qf     = (_Float16*)(ws + alloc((size_t)M * NH_ * HD_ * 2));       // 8 MB
  _Float16* kf     = (_Float16*)(ws + alloc((size_t)M * NH_ * HD_ * 2));       // 8 MB
  _Float16* vf     = (_Float16*)(ws + alloc((size_t)M * NH_ * V_D_ * 2));      // 8 MB
  float*    attn_o = q_raw;  // q_raw is dead after q_norm_rope; reuse

  dim3 gblk(256);

  // 1) q_raw = x @ Wq^T   (M=4096, N=1024, K=1024)
  gemm_f32_wmma<<<dim3(M / GBM, (NH_ * HD_) / GBN), gblk, 0, stream>>>(
      x, Wq, q_raw, M, NH_ * HD_, D_);
  // 2) RMSNorm + RoPE -> qf
  q_norm_rope<<<(M * NH_ + 255) / 256, gblk, 0, stream>>>(q_raw, q_norm_w, cosb, sinb, qf);
  // 3) kv_a = x @ Wkva^T  (N=144, guard tiles)
  gemm_f32_wmma<<<dim3(M / GBM, (R_ + ROPE_D_ + GBN - 1) / GBN), gblk, 0, stream>>>(
      x, Wkva, kv_a, M, R_ + ROPE_D_, D_);
  // 4) latent RMSNorm + k_rope RoPE
  kva_process<<<(M + 255) / 256, gblk, 0, stream>>>(kv_a, kva_w, cosb, sinb, latent, krope);
  // 5) kv_b = latent @ Wkvb^T  (N=1792, K=128)
  gemm_f32_wmma<<<dim3(M / GBM, (NH_ * (NOPE_D_ + V_D_)) / GBN), gblk, 0, stream>>>(
      latent, Wkvb, kv_b, M, NH_ * (NOPE_D_ + V_D_), R_);
  // 6) pack kf (k_nope | k_rope broadcast), vf
  pack_kv<<<(M * NH_ + 255) / 256, gblk, 0, stream>>>(kv_b, krope, kf, vf);
  // 7) causal flash attention -> attn_o (B,T,NH*64) f32
  mla_flash_attn<<<dim3(T_ / QT_, NH_, B_), dim3(128), 0, stream>>>(qf, kf, vf, attn_o);
  // 8) out = attn_o @ Wo^T  (M=4096, N=1024, K=1024)
  gemm_f32_wmma<<<dim3(M / GBM, D_ / GBN), gblk, 0, stream>>>(
      attn_o, Wo, out, M, D_, NH_ * V_D_);
}
